// W2Attention_7438883357211
// MI455X (gfx1250) — matrix operand order repaired and checked by reading, not compiled, not run on hardware
//
#include <hip/hip_runtime.h>
#include <cstdint>
#include <cstddef>

#define S_LEN 2048
#define HIDN  2048
#define NH    16
#define DH    128

typedef unsigned short ushort_t;
typedef __attribute__((ext_vector_type(16))) __bf16 v16bf;
typedef __attribute__((ext_vector_type(8)))  float  v8f;

struct U4 { unsigned int a, b, c, d; };
union Frag { v16bf v; U4 q[2]; };

// LDS row paddings (ushorts): keep 16B row alignment, avoid 64-bank-aligned strides
#define KLP 40    // K / sigmaK / P tiles: 20 dw stride -> conflict-free 16-lane b128
#define VMP 136   // V / B tiles: 68 dw stride -> 2-pass (minimum) 32-lane b128
#define ASP 40
#define BSP 136
#define CSP 129   // f32, +1 to de-align scalar epilogue traffic

static __device__ __forceinline__ ushort_t f2bf(float f) {
  unsigned u = __float_as_uint(f);
  u += 0x7fffu + ((u >> 16) & 1u);          // round-to-nearest-even
  return (ushort_t)(u >> 16);
}
static __device__ __forceinline__ unsigned pk2(float a, float b) {
  return (unsigned)f2bf(a) | ((unsigned)f2bf(b) << 16);
}
static __device__ __forceinline__ v8f wmma_bf16(v16bf a, v16bf b, v8f c) {
  // D = A(16x32 bf16) * B(32x16 bf16) + C(16x16 f32)
  return __builtin_amdgcn_wmma_f32_16x16x32_bf16(false, a, false, b, (short)0, c, false, false);
}

// ---- optional gfx1250 async global->LDS staging (guarded; falls back cleanly)
#if defined(__has_builtin)
#if __has_builtin(__builtin_amdgcn_global_load_async_to_lds_b128)
#define USE_ASYNC_LDS 1
#endif
#endif

#if defined(USE_ASYNC_LDS)
typedef __attribute__((__vector_size__(16))) int v4i_t;
typedef __attribute__((address_space(1))) v4i_t* gv4p;   // global int4*
typedef __attribute__((address_space(3))) v4i_t* lv4p;   // LDS int4*
static __device__ __forceinline__ void async_cp16(const ushort_t* g, ushort_t* l) {
  __builtin_amdgcn_global_load_async_to_lds_b128(
      (gv4p)(uintptr_t)g, (lv4p)(unsigned)(uintptr_t)l, 0, 0);
}
static __device__ __forceinline__ void wait_async0() {
#if __has_builtin(__builtin_amdgcn_s_wait_asynccnt)
  __builtin_amdgcn_s_wait_asynccnt(0);
#else
  asm volatile("s_wait_asynccnt 0x0" ::: "memory");
#endif
}
#endif

enum { MODE_NONE = 0, MODE_ROPE = 1, MODE_SIGMA = 2, MODE_OUT = 3 };

// ---------------------------------------------------------------------------
// Projection GEMM: Y[s,o] = sum_h X[s,h] * W[o,h]  (X: S x HID, W: HID x HID)
// Tile: 64 (M) x 128 (N) per WG, K-steps of 32, 8 waves each doing 32x32.
// ---------------------------------------------------------------------------
template <bool IN_F32, int MODE>
__global__ __launch_bounds__(256) void proj_kernel(
    const void* __restrict__ Xv, const float* __restrict__ W,
    ushort_t* __restrict__ Ybf, float* __restrict__ normOut,
    float* __restrict__ Yf32,
    const float* __restrict__ cosT, const float* __restrict__ sinT) {
  __shared__ __align__(16) ushort_t As[64][ASP];
  __shared__ __align__(16) ushort_t Bs[32][BSP];
  __shared__ __align__(16) float    Cs[64][CSP];
  __shared__ float np[64][4];

  const int t    = threadIdx.x;
  const int lane = t & 31;
  const int w    = t >> 5;
  const int hi   = lane >> 4;
  const int h8   = hi * 8;
  const int ln   = lane & 15;
  const int sB   = blockIdx.y * 64;
  const int oB   = blockIdx.x * 128;
  const int mW   = (w >> 2) * 32;
  const int nW   = (w & 3) * 32;

  v8f acc[2][2];
#pragma unroll
  for (int i = 0; i < 2; ++i)
#pragma unroll
    for (int j = 0; j < 2; ++j)
#pragma unroll
      for (int v = 0; v < 8; ++v) acc[i][j][v] = 0.0f;

  for (int kB = 0; kB < HIDN; kB += 32) {
    __syncthreads();
    // stage A tile (64 x 32) as bf16
    {
      const int r  = t >> 2;
      const int k0 = (t & 3) * 8;
      if (IN_F32) {
        const float* src = (const float*)Xv + (size_t)(sB + r) * HIDN + kB + k0;
        U4 u;
        u.a = pk2(src[0], src[1]); u.b = pk2(src[2], src[3]);
        u.c = pk2(src[4], src[5]); u.d = pk2(src[6], src[7]);
        *(U4*)&As[r][k0] = u;
      } else {
        const ushort_t* src = (const ushort_t*)Xv + (size_t)(sB + r) * HIDN + kB + k0;
        *(U4*)&As[r][k0] = *(const U4*)src;
      }
    }
    // stage B tile: Bs[kk][n] = W[oB+n][kB+kk] as bf16 (transposed store)
    {
      const int n  = t >> 1;
      const int k0 = (t & 1) * 16;
      const float* wr = W + (size_t)(oB + n) * HIDN + kB + k0;
#pragma unroll
      for (int j = 0; j < 16; ++j) Bs[k0 + j][n] = f2bf(wr[j]);
      if (kB + 32 < HIDN)
        __builtin_prefetch((const char*)(wr + 32), 0, 1);  // -> global_prefetch
    }
    __syncthreads();

    Frag a0, a1, b0, b1;
    a0.q[0] = *(const U4*)&As[mW + ln][h8];       a0.q[1] = *(const U4*)&As[mW + ln][16 + h8];
    a1.q[0] = *(const U4*)&As[mW + 16 + ln][h8];  a1.q[1] = *(const U4*)&As[mW + 16 + ln][16 + h8];
    for (int s = 0; s < 16; ++s) {
      const int kk = h8 + s + (s & 8);
      b0.v[s] = __builtin_bit_cast(__bf16, Bs[kk][nW + ln]);
      b1.v[s] = __builtin_bit_cast(__bf16, Bs[kk][nW + 16 + ln]);
    }

    acc[0][0] = wmma_bf16(a0.v, b0.v, acc[0][0]);
    acc[0][1] = wmma_bf16(a0.v, b1.v, acc[0][1]);
    acc[1][0] = wmma_bf16(a1.v, b0.v, acc[1][0]);
    acc[1][1] = wmma_bf16(a1.v, b1.v, acc[1][1]);
  }

  if (MODE == MODE_OUT) {
#pragma unroll
    for (int st = 0; st < 2; ++st)
#pragma unroll
      for (int ti = 0; ti < 2; ++ti)
#pragma unroll
        for (int v = 0; v < 8; ++v) {
          const int row = sB + mW + st * 16 + v + h8;
          const int col = oB + nW + ti * 16 + ln;
          Yf32[(size_t)row * HIDN + col] = acc[st][ti][v];
        }
  } else {
    // round-trip through LDS: this WG's N-tile (128) is exactly one head
#pragma unroll
    for (int st = 0; st < 2; ++st)
#pragma unroll
      for (int ti = 0; ti < 2; ++ti)
#pragma unroll
        for (int v = 0; v < 8; ++v)
          Cs[mW + st * 16 + v + h8][nW + ti * 16 + ln] = acc[st][ti][v];
    __syncthreads();

    const int head = blockIdx.x;
    const int r    = t & 63;
    const int c0   = (t >> 6) * 32;
    const int sg   = sB + r;
    float ss = 0.0f;
#pragma unroll 8
    for (int j = 0; j < 32; ++j) {
      const int d   = c0 + j;
      const float x = Cs[r][d];
      float val;
      if (MODE == MODE_ROPE) {
        const float pr = Cs[r][d ^ 64];
        const float cc = cosT[(size_t)sg * DH + d];
        const float sn = sinT[(size_t)sg * DH + d];
        val = x * cc + ((d < 64) ? -pr : pr) * sn;
      } else if (MODE == MODE_SIGMA) {
        val = __expf(0.5f * x);
      } else {
        val = x;
      }
      Ybf[((size_t)head * S_LEN + sg) * DH + d] = f2bf(val);
      ss += val * val;
    }
    np[r][t >> 6] = ss;
    __syncthreads();
    if (normOut && t < 64)
      normOut[(size_t)head * S_LEN + sB + t] = np[t][0] + np[t][1] + np[t][2] + np[t][3];
  }
}

// ---------------------------------------------------------------------------
// Flash-style W2 attention. One WG = 128 query rows of one head (8 waves x 16).
// ---------------------------------------------------------------------------
static __device__ __forceinline__ void scatterT(ushort_t (*dst)[KLP], int d0, int key,
                                                const ushort_t* src) {
  const U4 x0 = *(const U4*)src;
  const U4 x1 = *(const U4*)(src + 8);
  unsigned u[8] = {x0.a, x0.b, x0.c, x0.d, x1.a, x1.b, x1.c, x1.d};
#pragma unroll
  for (int j = 0; j < 8; ++j) {
    dst[d0 + 2 * j][key]     = (ushort_t)(u[j] & 0xffffu);
    dst[d0 + 2 * j + 1][key] = (ushort_t)(u[j] >> 16);
  }
}

__global__ __launch_bounds__(256) void w2attn_kernel(
    const ushort_t* __restrict__ qmu, const ushort_t* __restrict__ kmu,
    const ushort_t* __restrict__ vmu, const ushort_t* __restrict__ sq,
    const ushort_t* __restrict__ sk,  const ushort_t* __restrict__ vls,
    const float* __restrict__ qn, const float* __restrict__ kn,
    const float* __restrict__ sqn, const float* __restrict__ skn,
    const float* __restrict__ tau,
    ushort_t* __restrict__ attn_mu, ushort_t* __restrict__ attn_ls) {
  __shared__ __align__(16) ushort_t Kl[128][KLP];   // K_mu, d-major (transposed)
  __shared__ __align__(16) ushort_t Sl[128][KLP];   // sigma_k, d-major
  __shared__ __align__(16) ushort_t Vm[32][VMP];    // V_mu, key-major
  __shared__ __align__(16) ushort_t Vl[32][VMP];    // V_ls, key-major
  __shared__ __align__(16) ushort_t Pw[8][16][KLP]; // per-wave probs
  __shared__ float colt[32];

  const int t    = threadIdx.x;
  const int lane = t & 31;
  const int w    = t >> 5;
  const int hi   = lane >> 4;
  const int h8   = hi * 8;
  const int ln   = lane & 15;
  const int head = blockIdx.y;
  const int sB   = blockIdx.x * 128;
  const int sW   = sB + w * 16;
  const float invDen = 1.0f / (fabsf(tau[0]) + 1e-6f);

  // Q fragments (4 K-chunks of 32 over d=128), A layout: lane row = ln
  Frag qmF[4], qsF[4];
  {
    const ushort_t* qrow = qmu + ((size_t)head * S_LEN + (sW + ln)) * DH;
    const ushort_t* srow = sq  + ((size_t)head * S_LEN + (sW + ln)) * DH;
#pragma unroll
    for (int c = 0; c < 4; ++c) {
      qmF[c].q[0] = *(const U4*)(qrow + c * 32 + h8);
      qmF[c].q[1] = *(const U4*)(qrow + c * 32 + 16 + h8);
      qsF[c].q[0] = *(const U4*)(srow + c * 32 + h8);
      qsF[c].q[1] = *(const U4*)(srow + c * 32 + 16 + h8);
    }
  }
  float rt[8], mi[8], li[8];
#pragma unroll
  for (int v = 0; v < 8; ++v) {
    const int s = sW + v + h8;
    rt[v] = qn[(size_t)head * S_LEN + s] + sqn[(size_t)head * S_LEN + s];
    mi[v] = -1e30f;
    li[v] = 0.0f;
  }
  v8f omu[8], ols[8];
#pragma unroll
  for (int dt = 0; dt < 8; ++dt)
#pragma unroll
    for (int v = 0; v < 8; ++v) { omu[dt][v] = 0.0f; ols[dt][v] = 0.0f; }

  for (int kb = 0; kb < S_LEN; kb += 32) {
    __syncthreads();
    {
      const int key = t >> 3;
      const int d0  = (t & 7) * 16;
      const size_t base = ((size_t)head * S_LEN + kb + key) * DH + d0;
      scatterT(Kl, d0, key, kmu + base);
      scatterT(Sl, d0, key, sk + base);
#if defined(USE_ASYNC_LDS)
      async_cp16(vmu + base,     &Vm[key][d0]);
      async_cp16(vmu + base + 8, &Vm[key][d0 + 8]);
      async_cp16(vls + base,     &Vl[key][d0]);
      async_cp16(vls + base + 8, &Vl[key][d0 + 8]);
#else
      *(U4*)&Vm[key][d0]     = *(const U4*)(vmu + base);
      *(U4*)&Vm[key][d0 + 8] = *(const U4*)(vmu + base + 8);
      *(U4*)&Vl[key][d0]     = *(const U4*)(vls + base);
      *(U4*)&Vl[key][d0 + 8] = *(const U4*)(vls + base + 8);
#endif
      if (t < 32)
        colt[t] = kn[(size_t)head * S_LEN + kb + t] + skn[(size_t)head * S_LEN + kb + t];
    }
#if defined(USE_ASYNC_LDS)
    wait_async0();
#endif
    __syncthreads();

    // scores for two 16-key tiles; hoist all 8 B-fragments, then WMMA chain
    float sc[2][8];
#pragma unroll
    for (int n = 0; n < 2; ++n) {
      Frag bK[4], bS[4];
#pragma unroll
      for (int c = 0; c < 4; ++c) {
        for (int s = 0; s < 16; ++s) {
          const int kk = c * 32 + h8 + s + (s & 8);
          bK[c].v[s] = __builtin_bit_cast(__bf16, Kl[kk][n * 16 + ln]);
          bS[c].v[s] = __builtin_bit_cast(__bf16, Sl[kk][n * 16 + ln]);
        }
      }
      v8f a;
#pragma unroll
      for (int v = 0; v < 8; ++v) a[v] = 0.0f;
#pragma unroll
      for (int c = 0; c < 4; ++c) a = wmma_bf16(qmF[c].v, bK[c].v, a);
#pragma unroll
      for (int c = 0; c < 4; ++c) a = wmma_bf16(qsF[c].v, bS[c].v, a);
      const float ct = colt[n * 16 + ln];
#pragma unroll
      for (int v = 0; v < 8; ++v) sc[n][v] = (2.0f * a[v] - rt[v] - ct) * invDen;
    }

    // online softmax (row = v + 8*hi lives across the 16-lane half)
    float mb[8];
#pragma unroll
    for (int v = 0; v < 8; ++v) mb[v] = fmaxf(sc[0][v], sc[1][v]);
#pragma unroll
    for (int off = 1; off <= 8; off <<= 1)
#pragma unroll
      for (int v = 0; v < 8; ++v) mb[v] = fmaxf(mb[v], __shfl_xor(mb[v], off, 32));

    float al[8], p0[8], p1[8], rs[8];
#pragma unroll
    for (int v = 0; v < 8; ++v) {
      const float mn = fmaxf(mi[v], mb[v]);
      al[v] = __expf(mi[v] - mn);
      mi[v] = mn;
      p0[v] = __expf(sc[0][v] - mn);
      p1[v] = __expf(sc[1][v] - mn);
      rs[v] = p0[v] + p1[v];
    }
#pragma unroll
    for (int off = 1; off <= 8; off <<= 1)
#pragma unroll
      for (int v = 0; v < 8; ++v) rs[v] += __shfl_xor(rs[v], off, 32);
#pragma unroll
    for (int v = 0; v < 8; ++v) li[v] = li[v] * al[v] + rs[v];
#pragma unroll
    for (int dt = 0; dt < 8; ++dt)
#pragma unroll
      for (int v = 0; v < 8; ++v) { omu[dt][v] *= al[v]; ols[dt][v] *= al[v]; }

    // C-layout probs -> LDS -> A-layout fragment
#pragma unroll
    for (int v = 0; v < 8; ++v) {
      Pw[w][v + h8][ln]      = f2bf(p0[v]);
      Pw[w][v + h8][16 + ln] = f2bf(p1[v]);
    }
    __syncthreads();

    Frag pF;
    pF.q[0] = *(const U4*)&Pw[w][ln][h8];
    pF.q[1] = *(const U4*)&Pw[w][ln][16 + h8];
#pragma unroll
    for (int dt = 0; dt < 8; ++dt) {
      Frag bv, bl;
      for (int s = 0; s < 16; ++s) {
        const int kk = h8 + s + (s & 8);
        bv.v[s] = __builtin_bit_cast(__bf16, Vm[kk][dt * 16 + ln]);
        bl.v[s] = __builtin_bit_cast(__bf16, Vl[kk][dt * 16 + ln]);
      }
      omu[dt] = wmma_bf16(pF.v, bv.v, omu[dt]);
      ols[dt] = wmma_bf16(pF.v, bl.v, ols[dt]);
    }
  }

  // finalize: O /= l, write bf16 [s][head*128+d] (ready for Wo GEMM)
  float inv[8];
#pragma unroll
  for (int v = 0; v < 8; ++v) inv[v] = 1.0f / li[v];
#pragma unroll
  for (int dt = 0; dt < 8; ++dt)
#pragma unroll
    for (int v = 0; v < 8; ++v) {
      const int s   = sW + v + h8;
      const int col = head * DH + dt * 16 + ln;
      attn_mu[(size_t)s * HIDN + col] = f2bf(omu[dt][v] * inv[v]);
      attn_ls[(size_t)s * HIDN + col] = f2bf(ols[dt][v] * inv[v]);
    }
}

// ---------------------------------------------------------------------------
extern "C" void kernel_launch(void* const* d_in, const int* in_sizes, int n_in,
                              void* d_out, int out_size, void* d_ws, size_t ws_size,
                              hipStream_t stream) {
  (void)in_sizes; (void)n_in; (void)out_size; (void)ws_size;
  const float* hmu  = (const float*)d_in[0];
  const float* hls  = (const float*)d_in[1];
  const float* cosT = (const float*)d_in[2];
  const float* sinT = (const float*)d_in[3];
  const float* Wq   = (const float*)d_in[4];
  const float* Wk   = (const float*)d_in[5];
  const float* Wv   = (const float*)d_in[6];
  const float* Wo   = (const float*)d_in[7];
  const float* Wqs  = (const float*)d_in[8];
  const float* Wks  = (const float*)d_in[9];
  const float* Wvs  = (const float*)d_in[10];
  const float* Wos  = (const float*)d_in[11];
  const float* tau  = (const float*)d_in[12];

  const size_t nHSD = (size_t)NH * S_LEN * DH;   // 4,194,304 elements
  ushort_t* qmu = (ushort_t*)d_ws;
  ushort_t* kmu = qmu + 1 * nHSD;
  ushort_t* vmu = qmu + 2 * nHSD;
  ushort_t* sqb = qmu + 3 * nHSD;
  ushort_t* skb = qmu + 4 * nHSD;
  ushort_t* vlb = qmu + 5 * nHSD;
  ushort_t* amu = qmu + 6 * nHSD;                // attn out mu  [S][HID] bf16
  ushort_t* als = qmu + 7 * nHSD;                // attn out ls
  float* qn  = (float*)(qmu + 8 * nHSD);
  float* kn  = qn + (size_t)NH * S_LEN;
  float* sqn = kn + (size_t)NH * S_LEN;
  float* skn = sqn + (size_t)NH * S_LEN;

  float* out_mu = (float*)d_out;
  float* out_ls = out_mu + (size_t)S_LEN * HIDN;

  dim3 bG(256);
  dim3 gG(HIDN / 128, S_LEN / 64);  // (16, 32)
  proj_kernel<true, MODE_ROPE><<<gG, bG, 0, stream>>>(hmu, Wq,  qmu, qn,  nullptr, cosT, sinT);
  proj_kernel<true, MODE_ROPE><<<gG, bG, 0, stream>>>(hmu, Wk,  kmu, kn,  nullptr, cosT, sinT);
  proj_kernel<true, MODE_NONE><<<gG, bG, 0, stream>>>(hmu, Wv,  vmu, nullptr, nullptr, nullptr, nullptr);
  proj_kernel<true, MODE_SIGMA><<<gG, bG, 0, stream>>>(hls, Wqs, sqb, sqn, nullptr, nullptr, nullptr);
  proj_kernel<true, MODE_SIGMA><<<gG, bG, 0, stream>>>(hls, Wks, skb, skn, nullptr, nullptr, nullptr);
  proj_kernel<true, MODE_NONE><<<gG, bG, 0, stream>>>(hls, Wvs, vlb, nullptr, nullptr, nullptr, nullptr);

  dim3 gA(S_LEN / 128, NH);         // (16, 16)
  w2attn_kernel<<<gA, bG, 0, stream>>>(qmu, kmu, vmu, sqb, skb, vlb,
                                       qn, kn, sqn, skn, tau, amu, als);

  proj_kernel<false, MODE_OUT><<<gG, bG, 0, stream>>>(amu, Wo,  nullptr, nullptr, out_mu, nullptr, nullptr);
  proj_kernel<false, MODE_OUT><<<gG, bG, 0, stream>>>(als, Wos, nullptr, nullptr, out_ls, nullptr, nullptr);
}
